// MyVectorQuantizer_45157286150844
// MI455X (gfx1250) — matrix operand order repaired and checked by reading, not compiled, not run on hardware
//
#include <hip/hip_runtime.h>
#include <hip/hip_bf16.h>
#include <math.h>

// ---------------------------------------------------------------------------
// VQ (cosine) quantizer for MI455X / gfx1250.
//   x:  [64,8,128,256] f32  -> 65536 rows of dim 256
//   cb: [1024,256] f32
// Outputs (concatenated float, in return order):
//   [0]                loss
//   [1 .. 1+16777216)  quantized_st = cbn[idx]   (65536 x 256)
//   [16777217]         perplexity
//   [16777218 .. +262144) codebook passthrough (raw)
//   [17039362 .. +65536)  code indices (as float values)
// ---------------------------------------------------------------------------

typedef __bf16 v16bf __attribute__((ext_vector_type(16)));
typedef __bf16 v8bf  __attribute__((ext_vector_type(8)));
typedef float  v8f   __attribute__((ext_vector_type(8)));
typedef int    v4i   __attribute__((ext_vector_type(4)));

#define VQ_D        256
#define VQ_N        1024
#define VQ_ROWS     65536
#define VQ_QELEMS   16777216   // 65536*256
#define VQ_EPS      1e-12f

#define AS1 __attribute__((address_space(1)))
#define AS3 __attribute__((address_space(3)))

__device__ __forceinline__ float vq_wave_sum(float v) {
#pragma unroll
  for (int m = 16; m >= 1; m >>= 1) v += __shfl_xor(v, m, 32);
  return v;
}

// Async 16B global->LDS copy (ASYNCcnt path, bypasses VGPR round-trip).
// Builtin signature (probe-confirmed via diagnostic): (global v4i*, lds v4i*,
// imm offset, imm cpol).
__device__ __forceinline__ void vq_async_b128(const __bf16* g, void* lds) {
#if __has_builtin(__builtin_amdgcn_global_load_async_to_lds_b128)
  __builtin_amdgcn_global_load_async_to_lds_b128(
      (AS1 v4i*)(void*)g, (AS3 v4i*)lds, 0, 0);
#else
  const unsigned lds_off = (unsigned)(size_t)(AS3 __bf16*)lds;
  asm volatile("global_load_async_to_lds_b128 %0, %1, off"
               :: "v"(lds_off), "v"((unsigned long long)(size_t)g)
               : "memory");
#endif
}

__device__ __forceinline__ void vq_wait_async16() {
#if __has_builtin(__builtin_amdgcn_s_wait_asynccnt)
  __builtin_amdgcn_s_wait_asynccnt(16);
#else
  asm volatile("s_wait_asynccnt 0x10" ::: "memory");
#endif
}
__device__ __forceinline__ void vq_wait_async0() {
#if __has_builtin(__builtin_amdgcn_s_wait_asynccnt)
  __builtin_amdgcn_s_wait_asynccnt(0);
#else
  asm volatile("s_wait_asynccnt 0x0" ::: "memory");
#endif
}

// ---------------------------------------------------------------------------
// Kernel A: normalize codebook; emit fp32 copy, bf16 transposed copy [256][1024],
// raw passthrough; zero hist + sse accumulators (every launch -> deterministic).
// grid 128 x 256 threads (one wave per codebook row).
// ---------------------------------------------------------------------------
__global__ __launch_bounds__(256) void vq_prep_kernel(
    const float* __restrict__ cw, float* __restrict__ cbn,
    __bf16* __restrict__ cbT, unsigned* __restrict__ hist,
    float* __restrict__ gsse, float* __restrict__ outCb) {
  const int lane = threadIdx.x & 31;
  const int w    = threadIdx.x >> 5;
  const int n    = blockIdx.x * 8 + w;          // codebook row 0..1023

  const int gt = blockIdx.x * 256 + threadIdx.x;
  if (gt < VQ_N) hist[gt] = 0u;
  if (gt == 0)  *gsse = 0.0f;

  const float* xr = cw + (size_t)n * VQ_D + lane * 8;
  float v[8];
  float ss = 0.0f;
#pragma unroll
  for (int j = 0; j < 8; ++j) { v[j] = xr[j]; ss += v[j] * v[j]; }
  ss = vq_wave_sum(ss);
  const float scale = 1.0f / fmaxf(sqrtf(ss), VQ_EPS);

  float* cr = cbn   + (size_t)n * VQ_D + lane * 8;
  float* pr = outCb + (size_t)n * VQ_D + lane * 8;
#pragma unroll
  for (int j = 0; j < 8; ++j) {
    const float s = v[j] * scale;
    cr[j] = s;                                   // normalized fp32 (for gather)
    pr[j] = v[j];                                // raw passthrough output
    cbT[(size_t)(lane * 8 + j) * VQ_N + n] = (__bf16)s;  // transposed bf16 [k][n]
  }
}

// ---------------------------------------------------------------------------
// Kernel B: main GEMM + argmax + gather + SSE + histogram.
// grid 1024 blocks x 128 threads (4 waves). Each wave owns 16 x-rows.
// LDS: xn fp32 [64][260] + double-buffered codebook chunks bf16 [2][256][72].
// Codebook staged via async-to-LDS (ASYNCcnt), double buffered so the L2
// staging traffic is fully hidden behind the WMMA stream.
// ---------------------------------------------------------------------------
__global__ __launch_bounds__(128) void vq_main_kernel(
    const float* __restrict__ x, const float* __restrict__ cbn,
    const __bf16* __restrict__ cbT, float* __restrict__ outQ,
    float* __restrict__ outIdx, unsigned* __restrict__ hist,
    float* __restrict__ gsse) {
  __shared__ float  xn_lds[64][260];         // 66,560 B (4-bank lane skew)
  __shared__ __bf16 cb_lds[2][256][72];      // 73,728 B (row 144 B, 16B aligned)
  __shared__ int    idx_lds[4][16];

  const int lane    = threadIdx.x & 31;
  const int w       = threadIdx.x >> 5;
  const int rowBase = blockIdx.x * 64;
  const int wrow0   = w * 16;
  const int h       = lane >> 4;          // half-wave
  const int m       = lane & 15;

  // ---- Phase 1: normalize this wave's 16 rows into LDS (fp32) ----
  for (int r = 0; r < 16; ++r) {
    const int row = wrow0 + r;
    const float* xr = x + (size_t)(rowBase + row) * VQ_D + lane * 8;
    float v[8];
    float ss = 0.0f;
#pragma unroll
    for (int j = 0; j < 8; ++j) { v[j] = xr[j]; ss += v[j] * v[j]; }
    ss = vq_wave_sum(ss);
    const float scale = 1.0f / fmaxf(sqrtf(ss), VQ_EPS);
#pragma unroll
    for (int j = 0; j < 8; ++j) xn_lds[row][lane * 8 + j] = v[j] * scale;
  }

  // Prefetch codebook chunk 0 (64 codes x 256 k, 32 KB) into buffer 0.
  // Thread t stages rows k = t and k+128; 16 async b128 instr per wave.
  {
#pragma unroll
    for (int rr = 0; rr < 2; ++rr) {
      const int k = threadIdx.x + rr * 128;
      const __bf16* src = cbT + (size_t)k * VQ_N;   // chunk 0
#pragma unroll
      for (int j = 0; j < 8; ++j)
        vq_async_b128(src + j * 8, &cb_lds[0][k][j * 8]);
    }
  }
  __syncthreads();   // also publishes xn_lds

  // ---- Phase 2: build persistent A fragments (ISA 16-bit A 16x32 layout) ----
  // lane (h,m): element e<8 -> K = 8h+e ; e>=8 -> K = 16+8h+(e-8)
  v16bf afrag[8];
#pragma unroll
  for (int kb = 0; kb < 8; ++kb) {
#pragma unroll
    for (int e = 0; e < 16; ++e) {
      const int kk = (e < 8) ? (8 * h + e) : (16 + 8 * h + (e - 8));
      afrag[kb][e] = (__bf16)xn_lds[wrow0 + m][kb * 32 + kk];
    }
  }

  float maxv[8];
  int   maxi[8];
#pragma unroll
  for (int r = 0; r < 8; ++r) { maxv[r] = -__builtin_inff(); maxi[r] = 0; }

  // ---- Phase 3: 16 chunks of 64 codes, double-buffered async staging ----
  for (int c = 0; c < 16; ++c) {
    const int buf = c & 1;
    if (c + 1 < 16) {
      // Safe to overwrite buf (c+1)&1: barrier at end of chunk c-1 passed.
#pragma unroll
      for (int rr = 0; rr < 2; ++rr) {
        const int k = threadIdx.x + rr * 128;
        const __bf16* src = cbT + (size_t)k * VQ_N + (c + 1) * 64;
#pragma unroll
        for (int j = 0; j < 8; ++j)
          vq_async_b128(src + j * 8, &cb_lds[buf ^ 1][k][j * 8]);
      }
      vq_wait_async16();   // chunk c's 16 loads done (in-order); c+1 in flight
    } else {
      vq_wait_async0();
    }
    __syncthreads();

    for (int nt = 0; nt < 4; ++nt) {
      v8f acc = {0.f, 0.f, 0.f, 0.f, 0.f, 0.f, 0.f, 0.f};
#pragma unroll
      for (int kb = 0; kb < 8; ++kb) {
        // B 32x16: lane (h,m) holds column N = m; element e -> K = 8h+e (e<8), 16+8h+(e-8) (e>=8)
        v16bf bfrag;
#pragma unroll
        for (int e = 0; e < 16; ++e) {
          const int kk = (e < 8) ? (8 * h + e) : (16 + 8 * h + (e - 8));
          bfrag[e] = cb_lds[buf][kb * 32 + kk][nt * 16 + m];
        }
        acc = __builtin_amdgcn_wmma_f32_16x16x32_bf16(
            false, afrag[kb], false, bfrag, (short)0, acc, false, false);
      }
      // C layout: element r at lane -> row M = 8h+r, col N = m (+ tile base)
      const int nbase = c * 64 + nt * 16 + m;
#pragma unroll
      for (int r = 0; r < 8; ++r) {
        const float s = acc[r];
        if (s > maxv[r]) { maxv[r] = s; maxi[r] = nbase; }  // strict > keeps first
      }
    }
    __syncthreads();   // all waves done reading buf before it is refilled
  }

  // ---- Phase 4: argmax reduce across 16-lane halves (tie -> lowest index) ----
#pragma unroll
  for (int r = 0; r < 8; ++r) {
    float v = maxv[r];
    int   i = maxi[r];
#pragma unroll
    for (int mm = 1; mm < 16; mm <<= 1) {
      const float ov = __shfl_xor(v, mm, 32);
      const int   oi = __shfl_xor(i, mm, 32);
      if (ov > v || (ov == v && oi < i)) { v = ov; i = oi; }
    }
    maxv[r] = v;
    maxi[r] = i;
  }
  if (m == 0) {  // lanes 0 and 16: rows M = 8h + r
#pragma unroll
    for (int r = 0; r < 8; ++r) {
      const int M = h * 8 + r;
      idx_lds[w][M] = maxi[r];
      outIdx[rowBase + wrow0 + M] = (float)maxi[r];
      atomicAdd(&hist[maxi[r]], 1u);
    }
  }
  __syncthreads();

  // ---- Phase 5: gather fp32 codebook rows, write quantized, accumulate SSE ----
  float sse = 0.0f;
  for (int mrow = 0; mrow < 16; ++mrow) {
    const int idx = idx_lds[w][mrow];
    const float* q   = cbn + (size_t)idx * VQ_D + lane * 8;
    float* outp = outQ + (size_t)(rowBase + wrow0 + mrow) * VQ_D + lane * 8;
    const float* xnp = &xn_lds[wrow0 + mrow][lane * 8];
#pragma unroll
    for (int j = 0; j < 8; ++j) {
      const float qq = q[j];
      const float d  = qq - xnp[j];
      sse += d * d;
      outp[j] = qq;    // scalar b32 stores: out region starts at odd float offset
    }
  }
  sse = vq_wave_sum(sse);
  if (lane == 0) atomicAdd(gsse, sse);
}

// ---------------------------------------------------------------------------
// Kernel C: loss + perplexity. 1 block x 1024 threads.
// ---------------------------------------------------------------------------
__global__ __launch_bounds__(1024) void vq_final_kernel(
    const unsigned* __restrict__ hist, const float* __restrict__ gsse,
    float* __restrict__ outLoss, float* __restrict__ outPerp) {
  __shared__ float red[32];
  const int t = threadIdx.x;
  const float p = (float)hist[t] * (1.0f / (float)VQ_ROWS);
  float e = p * logf(p + 1e-10f);
#pragma unroll
  for (int mm = 16; mm >= 1; mm >>= 1) e += __shfl_xor(e, mm, 32);
  if ((t & 31) == 0) red[t >> 5] = e;
  __syncthreads();
  if (t < 32) {
    float v = red[t];
#pragma unroll
    for (int mm = 16; mm >= 1; mm >>= 1) v += __shfl_xor(v, mm, 32);
    if (t == 0) {
      *outPerp = expf(-v);
      // loss = q_latent + 0.25 * e_latent, both == mean((q - xn)^2)
      *outLoss = 1.25f * (*gsse) * (1.0f / (float)VQ_QELEMS);
    }
  }
}

// ---------------------------------------------------------------------------
extern "C" void kernel_launch(void* const* d_in, const int* in_sizes, int n_in,
                              void* d_out, int out_size, void* d_ws, size_t ws_size,
                              hipStream_t stream) {
  const float* x  = (const float*)d_in[0];   // 65536 x 256
  const float* cw = (const float*)d_in[1];   // 1024 x 256
  float* out = (float*)d_out;

  const size_t Q_OFF    = 1;
  const size_t PERP_OFF = (size_t)1 + VQ_QELEMS;            // 16777217
  const size_t CB_OFF   = PERP_OFF + 1;                     // 16777218
  const size_t IDX_OFF  = CB_OFF + (size_t)VQ_N * VQ_D;     // 17039362

  char* ws = (char*)d_ws;
  float*    cbn  = (float*)ws;                        // 1,048,576 B
  __bf16*   cbT  = (__bf16*)(ws + 1048576);           //   524,288 B
  unsigned* hist = (unsigned*)(ws + 1572864);         //     4,096 B
  float*    gsse = (float*)(ws + 1576960);            //         4 B

  vq_prep_kernel<<<128, 256, 0, stream>>>(cw, cbn, cbT, hist, gsse, out + CB_OFF);
  vq_main_kernel<<<1024, 128, 0, stream>>>(x, cbn, cbT, out + Q_OFF,
                                           out + IDX_OFF, hist, gsse);
  vq_final_kernel<<<1, 1024, 0, stream>>>(hist, gsse, out, out + PERP_OFF);
}
